// Nonlocal_block_12610023981067
// MI455X (gfx1250) — matrix operand order repaired and checked by reading, not compiled, not run on hardware
//
#include <hip/hip_runtime.h>
#include <hip/hip_bf16.h>

typedef __attribute__((ext_vector_type(16))) _Float16 v16h;
typedef __attribute__((ext_vector_type(8)))  float    v8f;

union H8 {
  uint4 v;
  _Float16 h[8];
};

// ---------------------------------------------------------------------------
// WMMA fragment helpers (wave32, v_wmma_f32_16x16x32_f16; contiguous k = 16*hi + h in both A and B)
// Both loaders read 16 contiguous halves per lane (2 x ds_load_b128).
// ---------------------------------------------------------------------------

__device__ __forceinline__ int a_kmap(int h, int hi) {
  // A 16-bit 16x32: lanes 0-15 row M=l: K=0..15
  //                 lanes 16-31 row M=l: K=16..31
  return hi * 16 + h;
}

// logical A[row][k] = src[row*ld + k]   (16 contiguous halves per lane)
__device__ __forceinline__ v16h frag_a(const _Float16* src, int ld) {
  int l = threadIdx.x & 15, hi = (threadIdx.x >> 4) & 1;
  v16h a;
#pragma unroll
  for (int h = 0; h < 16; ++h) a[h] = src[l * ld + a_kmap(h, hi)];
  return a;
}

// logical B[k][n] = src[n*ld + k]       (16 contiguous halves per lane)
__device__ __forceinline__ v16h frag_b_trans(const _Float16* src, int ld) {
  int l = threadIdx.x & 15, hi = (threadIdx.x >> 4) & 1;
  v16h b;
#pragma unroll
  for (int h = 0; h < 16; ++h) b[h] = src[l * ld + hi * 16 + h];
  return b;
}

__device__ __forceinline__ v8f wmma16(v16h a, v16h b, v8f c) {
  return __builtin_amdgcn_wmma_f32_16x16x32_f16(false, a, false, b, (short)0, c,
                                                false, false);
}

// Problem constants
#define BB 8
#define CC 128
#define CB 64
#define NN 4096

// ---------------------------------------------------------------------------
// Kernel 1: theta/phi/g = W[64x128] @ xf[128xN].
//   theta, phi stored TRANSPOSED in global ([b][n][k]) -> fragment-friendly.
//   g stored [b][k][n] (its B-read in kernel 2 is already contiguous).
// grid = B * N/64, 256 threads (8 waves).
// ---------------------------------------------------------------------------
__global__ __launch_bounds__(256) void nl_proj_kernel(
    const float* __restrict__ x, const float* __restrict__ wth,
    const float* __restrict__ wph, const float* __restrict__ wg,
    _Float16* __restrict__ th_t, _Float16* __restrict__ ph_t,
    _Float16* __restrict__ g) {
  __shared__ _Float16 x_t[64][136];  // x transposed: [n][c]
  __shared__ _Float16 w_s[64][136];  // weight row-major: [k][c]

  int b = blockIdx.x >> 6;
  int n0 = (blockIdx.x & 63) * 64;
  int t = threadIdx.x;
  int w = t >> 5, l = t & 15, hi = (t >> 4) & 1;

  {  // stage x block [128 c][64 n] transposed into x_t[n][c] (f16 scatter)
    int c = t >> 1, seg = (t & 1) * 32;
    const float* src = x + ((size_t)b * CC + c) * NN + n0 + seg;
#pragma unroll
    for (int j = 0; j < 32; ++j) x_t[seg + j][c] = (_Float16)src[j];
  }

  const float* wsrc[3] = {wth, wph, wg};

  for (int p = 0; p < 3; ++p) {
    __syncthreads();
    {  // stage weight [64 k][128 c] row-major as f16
      int r = t >> 2, seg = (t & 3) * 32;
      const float* src = wsrc[p] + r * 128 + seg;
#pragma unroll
      for (int j = 0; j < 32; ++j) w_s[r][seg + j] = (_Float16)src[j];
    }
    __syncthreads();
#pragma unroll
    for (int i = 0; i < 2; ++i) {  // 16 tiles of 16x16 over 8 waves
      int tile = w * 2 + i, rt = tile >> 2, ct = tile & 3;
      v8f acc = {};
#pragma unroll
      for (int kk = 0; kk < 4; ++kk)
        acc = wmma16(frag_a(&w_s[rt * 16][kk * 32], 136),
                     frag_b_trans(&x_t[ct * 16][kk * 32], 136), acc);
      // C layout: lane -> (row kb = rt*16+hi*8+r, col n = ct*16+l)
      if (p < 2) {
        // transposed store [b][n][k]: 8 contiguous halves along k -> 1x b128
        _Float16* dst = (p == 0 ? th_t : ph_t) +
                        ((size_t)b * NN + n0 + ct * 16 + l) * CB + rt * 16 +
                        hi * 8;
        H8 o;
#pragma unroll
        for (int r = 0; r < 8; ++r) o.h[r] = (_Float16)acc[r];
        *(uint4*)dst = o.v;
      } else {
        // g: [b][k][n], strided along k
        _Float16* dst =
            g + ((size_t)b * CB + rt * 16 + hi * 8) * NN + n0 + ct * 16 + l;
#pragma unroll
        for (int r = 0; r < 8; ++r) dst[(size_t)r * NN] = (_Float16)acc[r];
      }
    }
  }
}

// ---------------------------------------------------------------------------
// Kernel 2: flash attention.  grid = B * N/128, 256 threads (8 waves).
// Wave w owns query rows n0 + w*16..+15 (128 per block); sweeps 4096 keys in
// blocks of 64 with register double-buffering of the phi/g staging loads so
// the HBM latency hides under the previous block's WMMAs.
// ---------------------------------------------------------------------------
__global__ __launch_bounds__(256) void nl_attn_kernel(
    const _Float16* __restrict__ th_t, const _Float16* __restrict__ ph_t,
    const _Float16* __restrict__ g, _Float16* __restrict__ y) {
  __shared__ _Float16 th_s[128][72];   // [n][k]
  __shared__ _Float16 ph_s[64][72];    // [m][k]
  __shared__ _Float16 g_s[64][72];     // [k][m]
  __shared__ _Float16 p_s[8][16][72];  // per-wave P [n][m]

  int b = blockIdx.x >> 5;  // 32 query blocks of 128 per batch
  int n0 = (blockIdx.x & 31) * 128;
  int t = threadIdx.x;
  int w = t >> 5, l = t & 15, hi = (t >> 4) & 1;

  {  // stage theta block once: [128 n][64 k], rows contiguous in k
    int row = t >> 1, seg = (t & 1) * 32;
    const uint4* src =
        (const uint4*)(th_t + ((size_t)b * NN + n0 + row) * CB + seg);
    uint4* dst = (uint4*)&th_s[row][seg];
    dst[0] = src[0];
    dst[1] = src[1];
  }

  // phi/g staging coords: 64 rows x 4 segs of 16 halves over 256 threads
  int srow = t >> 2, sseg = (t & 3) * 16;
  const uint4* phsrc = (const uint4*)(ph_t + ((size_t)b * NN + srow) * CB + sseg);
  const uint4* gsrc = (const uint4*)(g + ((size_t)b * CB + srow) * NN + sseg);
  // preload key block 0 into registers
  uint4 pr0 = phsrc[0], pr1 = phsrc[1];
  uint4 gr0 = gsrc[0], gr1 = gsrc[1];

  v8f acc[4] = {};       // y accumulator: 16 n x 64 k, 4 C-tiles
  float m_i[8], l_i[8];  // per C-row running max / lane-partial sum
#pragma unroll
  for (int r = 0; r < 8; ++r) {
    m_i[r] = -__builtin_inff();
    l_i[r] = 0.0f;
  }

  for (int mb = 0; mb < 64; ++mb) {
    __syncthreads();  // previous block's compute done reading ph_s/g_s
    {                 // commit double-buffered registers to LDS
      uint4* dp = (uint4*)&ph_s[srow][sseg];
      dp[0] = pr0;
      dp[1] = pr1;
      uint4* dg = (uint4*)&g_s[srow][sseg];
      dg[0] = gr0;
      dg[1] = gr1;
    }
    if (mb + 1 < 64) {  // issue next block's loads; wait lands next iteration
      int moff = (mb + 1) * 64;
      pr0 = phsrc[moff * CB / 8];
      pr1 = phsrc[moff * CB / 8 + 1];
      gr0 = gsrc[moff / 8];
      gr1 = gsrc[moff / 8 + 1];
      if (mb + 2 < 64) {  // warm L2 two blocks ahead (global_prefetch_b8)
        __builtin_prefetch(ph_t + ((size_t)b * NN + moff + 64 + srow) * CB + sseg,
                           0, 1);
        __builtin_prefetch(g + ((size_t)b * CB + srow) * NN + moff + 64 + sseg,
                           0, 1);
      }
    }
    __syncthreads();

    // S = theta^T @ phi : 4 tiles of 16x16, K=64 (2 WMMAs each)
    v8f st[4];
#pragma unroll
    for (int ct = 0; ct < 4; ++ct) {
      v8f s = {};
      s = wmma16(frag_a(&th_s[w * 16][0], 72),
                 frag_b_trans(&ph_s[ct * 16][0], 72), s);
      s = wmma16(frag_a(&th_s[w * 16][32], 72),
                 frag_b_trans(&ph_s[ct * 16][32], 72), s);
      st[ct] = s;
    }

    // online softmax: a C-layout row spans the 16-lane half, so shfl_xor over
    // masks 1,2,4,8 reduces each row within its half.
    float corr[8];
#pragma unroll
    for (int r = 0; r < 8; ++r) {
      float mx = fmaxf(fmaxf(st[0][r], st[1][r]), fmaxf(st[2][r], st[3][r]));
#pragma unroll
      for (int sft = 1; sft <= 8; sft <<= 1)
        mx = fmaxf(mx, __shfl_xor(mx, sft, 32));
      float mnew = fmaxf(m_i[r], mx);
      corr[r] = __expf(m_i[r] - mnew);
      m_i[r] = mnew;
    }
#pragma unroll
    for (int ct = 0; ct < 4; ++ct)
#pragma unroll
      for (int r = 0; r < 8; ++r) st[ct][r] = __expf(st[ct][r] - m_i[r]);
#pragma unroll
    for (int r = 0; r < 8; ++r) {
      float ps = st[0][r] + st[1][r] + st[2][r] + st[3][r];
      l_i[r] = l_i[r] * corr[r] + ps;  // lane-partial; reduced at the end
#pragma unroll
      for (int kt = 0; kt < 4; ++kt) acc[kt][r] *= corr[r];
    }

    // C-layout -> A-fragment layout via LDS round trip (b16 scatter)
#pragma unroll
    for (int ct = 0; ct < 4; ++ct)
#pragma unroll
      for (int r = 0; r < 8; ++r)
        p_s[w][hi * 8 + r][ct * 16 + l] = (_Float16)st[ct][r];
    __syncthreads();

    // y += P @ g^T : 4 k-tiles, K=64 over m (2 WMMAs each); B[m][k]=g_s[k][m]
#pragma unroll
    for (int kt = 0; kt < 4; ++kt) {
      acc[kt] = wmma16(frag_a(&p_s[w][0][0], 72),
                       frag_b_trans(&g_s[kt * 16][0], 72), acc[kt]);
      acc[kt] = wmma16(frag_a(&p_s[w][0][32], 72),
                       frag_b_trans(&g_s[kt * 16][32], 72), acc[kt]);
    }
  }

  // finish row sums across the lane-half, normalize, store y ([b][k][n], f16)
#pragma unroll
  for (int r = 0; r < 8; ++r)
#pragma unroll
    for (int sft = 1; sft <= 8; sft <<= 1)
      l_i[r] += __shfl_xor(l_i[r], sft, 32);

#pragma unroll
  for (int kt = 0; kt < 4; ++kt) {
    _Float16* dst =
        y + ((size_t)b * CB + kt * 16 + l) * NN + n0 + w * 16 + hi * 8;
    H8 o;
#pragma unroll
    for (int r = 0; r < 8; ++r) o.h[r] = (_Float16)(acc[kt][r] / l_i[r]);
    *(uint4*)dst = o.v;
  }
}

// ---------------------------------------------------------------------------
// Kernel 3: out = w_last[128x64] @ y + x.  grid = B * N/64, 256 threads.
// ---------------------------------------------------------------------------
__global__ __launch_bounds__(256) void nl_out_kernel(
    const _Float16* __restrict__ y, const float* __restrict__ wlast,
    const float* __restrict__ x, float* __restrict__ out) {
  __shared__ _Float16 wl_s[128][72];  // [c][k] row-major
  __shared__ _Float16 y_t[64][72];    // y transposed: [n][k]

  int b = blockIdx.x >> 6;
  int n0 = (blockIdx.x & 63) * 64;
  int t = threadIdx.x;
  int w = t >> 5, l = t & 15, hi = (t >> 4) & 1;

  {  // stage w_last [128 c][64 k] row-major
    int c = t >> 1, seg = (t & 1) * 32;
    const float* src = wlast + c * 64 + seg;
#pragma unroll
    for (int j = 0; j < 32; ++j) wl_s[c][seg + j] = (_Float16)src[j];
  }
  {  // stage y block [64 k][64 n] transposed into y_t[n][k] (b16 scatter)
    int k = t >> 2, seg = (t & 3) * 16;
    const uint4* src = (const uint4*)(y + ((size_t)b * CB + k) * NN + n0 + seg);
    H8 u0, u1;
    u0.v = src[0];
    u1.v = src[1];
#pragma unroll
    for (int j = 0; j < 8; ++j) {
      y_t[seg + j][k] = u0.h[j];
      y_t[seg + 8 + j][k] = u1.h[j];
    }
  }
  __syncthreads();

#pragma unroll
  for (int ct = 0; ct < 4; ++ct) {  // 8 row-tiles x 4 col-tiles over 8 waves
    int rt = w;
    v8f acc = {};
    acc = wmma16(frag_a(&wl_s[rt * 16][0], 72),
                 frag_b_trans(&y_t[ct * 16][0], 72), acc);
    acc = wmma16(frag_a(&wl_s[rt * 16][32], 72),
                 frag_b_trans(&y_t[ct * 16][32], 72), acc);
    size_t base = ((size_t)b * CC + rt * 16 + hi * 8) * NN + n0 + ct * 16 + l;
#pragma unroll
    for (int r = 0; r < 8; ++r) {
      size_t off = base + (size_t)r * NN;
      out[off] = acc[r] + x[off];
    }
  }
}

// ---------------------------------------------------------------------------
extern "C" void kernel_launch(void* const* d_in, const int* in_sizes, int n_in,
                              void* d_out, int out_size, void* d_ws,
                              size_t ws_size, hipStream_t stream) {
  const float* x = (const float*)d_in[0];
  const float* wth = (const float*)d_in[1];
  const float* wph = (const float*)d_in[2];
  const float* wg = (const float*)d_in[3];
  const float* wlast = (const float*)d_in[4];
  float* out = (float*)d_out;

  // workspace: theta^T/phi^T/g/y, each 8*64*4096 f16 = 4 MB (16 MB total)
  const size_t elems = (size_t)BB * CB * NN;
  _Float16* th_t = (_Float16*)d_ws;  // [b][n][k]
  _Float16* ph_t = th_t + elems;     // [b][m][k]
  _Float16* g = ph_t + elems;        // [b][k][m]
  _Float16* y = g + elems;           // [b][k][n]

  nl_proj_kernel<<<dim3(BB * (NN / 64)), 256, 0, stream>>>(x, wth, wph, wg,
                                                           th_t, ph_t, g);
  nl_attn_kernel<<<dim3(BB * (NN / 128)), 256, 0, stream>>>(th_t, ph_t, g, y);
  nl_out_kernel<<<dim3(BB * (NN / 64)), 256, 0, stream>>>(y, wlast, x, out);
}
